// MultiHeadLatentAttention_19258633355609
// MI455X (gfx1250) — matrix operand order repaired and checked by reading, not compiled, not run on hardware
//
#include <hip/hip_runtime.h>

// ---------------------------------------------------------------------------
// MLA forward for gfx1250 (MI455X): bf16 WMMA GEMMs (async-LDS staged,
// double-buffered) + WMMA flash attention.
// ---------------------------------------------------------------------------

typedef __bf16 bf16;
typedef __attribute__((ext_vector_type(16))) __bf16 v16bf;
typedef __attribute__((ext_vector_type(8)))  __bf16 v8bf;
typedef __attribute__((ext_vector_type(8)))  float  v8f;

#define WMMA_BF16(a, b, c) \
  __builtin_amdgcn_wmma_f32_16x16x32_bf16(false, (a), false, (b), (short)0, (c), false, false)

// Model constants
#define D_MODEL 2048
#define D_HEAD  128
#define N_HEADS 16
#define D_LATENT 512
#define BATCH   2
#define SEQ     2048
#define ROWS    (BATCH * SEQ)           // 4096
#define ATTN_SCALE 0.08838834764831843f // 1/sqrt(128)

#define LDS_PAD_ROW 40                  // 32 k-halves + pad -> 80B row stride

// ---------------------------------------------------------------------------
// Fragment loaders (contiguous k order k = 16*(lane/16) + i in A and B, wave32)
// ---------------------------------------------------------------------------

// A-matrix 16x32 bf16: lane L holds row (L&15); element i -> k = (L>=16?16:0) + i
static __device__ __forceinline__ v16bf load_a_frag(const bf16* __restrict__ A, int lda,
                                                    int mBase, int kBase, int lane) {
  const int m    = mBase + (lane & 15);
  const int ksub = (lane >> 4) << 4;            // 0 or 16
  const bf16* p  = A + (size_t)m * lda + kBase + ksub;
  v8bf lo = *(const v8bf*)(p);                  // k = kBase+ksub .. +7
  v8bf hi = *(const v8bf*)(p + 8);              // k = kBase+ksub+8 .. +15
  v16bf a;
#pragma unroll
  for (int i = 0; i < 8; ++i) { a[i] = lo[i]; a[i + 8] = hi[i]; }
  return a;
}

// Async copy of 16 bytes global -> LDS (ASYNCcnt-tracked)
static __device__ __forceinline__ void async_copy_b128(const bf16* gsrc, bf16* ldst) {
  const unsigned loff = (unsigned)(uintptr_t)ldst;            // low 32 bits = LDS offset
  const unsigned long long ga = (unsigned long long)(uintptr_t)gsrc;
  asm volatile("global_load_async_to_lds_b128 %0, %1, off"
               :: "v"(loff), "v"(ga) : "memory");
}

// ---------------------------------------------------------------------------
// f32 -> bf16 conversion
// ---------------------------------------------------------------------------
__global__ __launch_bounds__(256) void cvt_f32_to_bf16(const float* __restrict__ src,
                                                       bf16* __restrict__ dst, int n) {
  int i = blockIdx.x * blockDim.x + threadIdx.x;
  if (i < n) dst[i] = (bf16)src[i];
}

// ---------------------------------------------------------------------------
// GEMM: Y = X[M,K] @ W[N,K]^T, bf16 in, f32 accumulate.
// Block = 256 threads (8 waves). Block tile: 256(M) x 64(N).
// Each wave: 32x64 tile = 2x4 WMMA D-tiles, 8 WMMAs per 32-wide K-step.
// W slab (64 rows x 32 k) staged in LDS via global_load_async_to_lds_b128,
// double-buffered; B fragments read from LDS (conflict-free, 80B row stride).
// Grid: (N/64, M/256).
// OUT mode: 0 = row-major [M,N]; 1 = [b,h,s,d]; 2 = [b,h,d,s] (for V^T).
// ---------------------------------------------------------------------------
template <int MODE, bool F32OUT>
__global__ __launch_bounds__(256) void gemm_bf16_wmma(const bf16* __restrict__ X,
                                                      const bf16* __restrict__ W,
                                                      void* __restrict__ Yv,
                                                      int M, int N, int K) {
  __shared__ bf16 Wslab[2][64 * LDS_PAD_ROW];

  const int tid   = threadIdx.x;
  const int lane  = tid & 31;
  const int wave  = tid >> 5;
  const int mBase = blockIdx.y * 256 + wave * 32;
  const int nBase = blockIdx.x * 64;
  const int half  = lane >> 4;
  const int nloc  = lane & 15;

  // cooperative slab copy: 64 rows x 4 segments of 8 halves (16B) = 256 threads
  const int cn  = tid >> 2;          // slab row 0..63
  const int ckh = (tid & 3) * 8;     // k-half segment 0,8,16,24
  const bf16* wrow = W + (size_t)(nBase + cn) * K + ckh;

  v8f acc[2][4];
#pragma unroll
  for (int i = 0; i < 2; ++i)
#pragma unroll
    for (int t = 0; t < 4; ++t)
#pragma unroll
      for (int r = 0; r < 8; ++r) acc[i][t][r] = 0.0f;

  // preload k=0 slab into buffer 0
  async_copy_b128(wrow, &Wslab[0][cn * LDS_PAD_ROW + ckh]);

  int cur = 0;
  for (int k = 0; k < K; k += 32) {
    const bool more = (k + 32) < K;
    if (more) {
      async_copy_b128(wrow + (k + 32), &Wslab[cur ^ 1][cn * LDS_PAD_ROW + ckh]);
      asm volatile("s_wait_asynccnt 0x1" ::: "memory");
    } else {
      asm volatile("s_wait_asynccnt 0x0" ::: "memory");
    }
    __syncthreads();                 // slab[cur] visible to all waves

    v16bf a0 = load_a_frag(X, K, mBase,      k, lane);
    v16bf a1 = load_a_frag(X, K, mBase + 16, k, lane);
    const bf16* bl = &Wslab[cur][0];
#pragma unroll
    for (int t = 0; t < 4; ++t) {
      // B frag: lane column n = t*16+nloc, 16 contiguous k-halves at kh = half*16
      const bf16* p = bl + (t * 16 + nloc) * LDS_PAD_ROW + half * 16;
      v8bf lo = *(const v8bf*)p;
      v8bf hi = *(const v8bf*)(p + 8);
      v16bf b;
#pragma unroll
      for (int i = 0; i < 8; ++i) { b[i] = lo[i]; b[i + 8] = hi[i]; }
      acc[0][t] = WMMA_BF16(a0, b, acc[0][t]);
      acc[1][t] = WMMA_BF16(a1, b, acc[1][t]);
    }
    __syncthreads();                 // all reads of slab[cur] done before overwrite
    cur ^= 1;
  }

#pragma unroll
  for (int i = 0; i < 2; ++i) {
#pragma unroll
    for (int t = 0; t < 4; ++t) {
#pragma unroll
      for (int r = 0; r < 8; ++r) {
        const int row = mBase + i * 16 + r + half * 8;  // = b*SEQ + s
        const int col = nBase + t * 16 + nloc;          // = h*D_HEAD + d
        const float v = acc[i][t][r];
        size_t idx;
        if (MODE == 0) {
          idx = (size_t)row * N + col;
        } else {
          const int b = row >> 11, s = row & (SEQ - 1);
          const int h = col >> 7,  d = col & (D_HEAD - 1);
          if (MODE == 1)
            idx = (((size_t)(b * N_HEADS + h)) * SEQ + s) * D_HEAD + d;
          else
            idx = (((size_t)(b * N_HEADS + h)) * D_HEAD + d) * SEQ + s;
        }
        if (F32OUT) ((float*)Yv)[idx] = v;
        else        ((bf16*)Yv)[idx]  = (bf16)v;
      }
    }
  }
}

// ---------------------------------------------------------------------------
// Flash attention, causal. One wave per (b,h,16-query block).
// Q,K: [b,h,s,d]; V: [b,h,d,s] (transposed); Ctx out: [b*s, h*d] bf16.
// ---------------------------------------------------------------------------
__global__ __launch_bounds__(256) void mla_flash_attn(const bf16* __restrict__ Q,
                                                      const bf16* __restrict__ Km,
                                                      const bf16* __restrict__ Vt,
                                                      bf16* __restrict__ Ctx) {
  __shared__ bf16 plds[8][16 * 32];             // per-wave 16x32 probability patch

  const int lane = threadIdx.x & 31;
  const int wave = threadIdx.x >> 5;
  const int wid  = blockIdx.x * 8 + wave;       // 0 .. 4095
  const int bh   = wid >> 7;                    // 0 .. 31
  const int qb   = wid & 127;                   // q block (of 16)
  const int b    = bh >> 4, h = bh & 15;

  const bf16* Qp = Q  + (size_t)bh * SEQ * D_HEAD;
  const bf16* Kp = Km + (size_t)bh * SEQ * D_HEAD;
  const bf16* Vp = Vt + (size_t)bh * D_HEAD * SEQ;

  const int mBase = qb * 16;
  const int half  = lane >> 4;
  const int nloc  = lane & 15;
  bf16* myp = plds[wave];

  v16bf qf[4];
#pragma unroll
  for (int kk = 0; kk < 4; ++kk) qf[kk] = load_a_frag(Qp, D_HEAD, mBase, kk * 32, lane);

  v8f acc[8];
#pragma unroll
  for (int t = 0; t < 8; ++t)
#pragma unroll
    for (int r = 0; r < 8; ++r) acc[t][r] = 0.0f;

  float mrow[8], lrow[8];
#pragma unroll
  for (int r = 0; r < 8; ++r) { mrow[r] = -1e30f; lrow[r] = 0.0f; }

  const int qEnd = mBase + 15;
  for (int kb = 0; kb <= qEnd; kb += 32) {
    v8f s0, s1;
#pragma unroll
    for (int r = 0; r < 8; ++r) { s0[r] = 0.0f; s1[r] = 0.0f; }
#pragma unroll
    for (int kk = 0; kk < 4; ++kk) {
      const bf16* p0 = Kp + (size_t)(kb + nloc) * D_HEAD + kk * 32 + half * 16;
      v16bf b0 = *(const v16bf*)p0;
      v16bf b1 = *(const v16bf*)(p0 + 16 * D_HEAD);
      s0 = WMMA_BF16(qf[kk], b0, s0);
      s1 = WMMA_BF16(qf[kk], b1, s1);
    }

#pragma unroll
    for (int r = 0; r < 8; ++r) {
      const int M    = r + half * 8;
      const int qpos = mBase + M;
      const int k0   = kb + nloc, k1 = k0 + 16;
      float f0 = (k0 <= qpos) ? s0[r] * ATTN_SCALE : -1e30f;
      float f1 = (k1 <= qpos) ? s1[r] * ATTN_SCALE : -1e30f;
      float rm = fmaxf(f0, f1);
#pragma unroll
      for (int off = 1; off < 16; off <<= 1) rm = fmaxf(rm, __shfl_xor(rm, off, 32));
      const float mnew  = fmaxf(mrow[r], rm);
      const float alpha = __expf(mrow[r] - mnew);
      const float p0 = __expf(f0 - mnew);
      const float p1 = __expf(f1 - mnew);
      float rs = p0 + p1;
#pragma unroll
      for (int off = 1; off < 16; off <<= 1) rs += __shfl_xor(rs, off, 32);
      lrow[r] = lrow[r] * alpha + rs;
      mrow[r] = mnew;
#pragma unroll
      for (int t = 0; t < 8; ++t) acc[t][r] *= alpha;
      myp[M * 32 + nloc]      = (bf16)p0;
      myp[M * 32 + nloc + 16] = (bf16)p1;
    }

    asm volatile("s_wait_dscnt 0x0" ::: "memory");   // intra-wave LDS RAW

    v16bf af;
    {
      v8bf lo = *(const v8bf*)&myp[nloc * 32 + half * 16];
      v8bf hi = *(const v8bf*)&myp[nloc * 32 + half * 16 + 8];
#pragma unroll
      for (int i = 0; i < 8; ++i) { af[i] = lo[i]; af[i + 8] = hi[i]; }
    }

#pragma unroll
    for (int t = 0; t < 8; ++t) {
      const bf16* pv = Vp + (size_t)(t * 16 + nloc) * SEQ + kb + half * 16;
      v16bf bv = *(const v16bf*)pv;
      acc[t] = WMMA_BF16(af, bv, acc[t]);
    }
  }

#pragma unroll
  for (int r = 0; r < 8; ++r) {
    const int M = r + half * 8;
    const int s = mBase + M;
    const float inv = 1.0f / lrow[r];
    const size_t rowoff = ((size_t)(b * SEQ) + s) * (N_HEADS * D_HEAD) + h * D_HEAD;
#pragma unroll
    for (int t = 0; t < 8; ++t)
      Ctx[rowoff + t * 16 + nloc] = (bf16)(acc[t][r] * inv);
  }
}

// ---------------------------------------------------------------------------
// Host launcher
// ---------------------------------------------------------------------------
extern "C" void kernel_launch(void* const* d_in, const int* in_sizes, int n_in,
                              void* d_out, int out_size, void* d_ws, size_t ws_size,
                              hipStream_t stream) {
  (void)in_sizes; (void)n_in; (void)out_size; (void)ws_size;

  const float* x  = (const float*)d_in[0];   // [2,2048,2048]
  const float* Wd = (const float*)d_in[1];   // [512,2048]
  const float* Wk = (const float*)d_in[2];   // [2048,512]
  const float* Wv = (const float*)d_in[3];   // [2048,512]
  const float* Wq = (const float*)d_in[4];   // [2048,2048]
  const float* Wo = (const float*)d_in[5];   // [2048,2048]
  float* out = (float*)d_out;

  // workspace layout (bf16 element offsets)
  bf16* ws = (bf16*)d_ws;
  bf16* x_bf   = ws;                         // 8388608
  bf16* Wd_bf  = x_bf  + 8388608;            // 1048576
  bf16* Wk_bf  = Wd_bf + 1048576;            // 1048576
  bf16* Wv_bf  = Wk_bf + 1048576;            // 1048576
  bf16* Wq_bf  = Wv_bf + 1048576;            // 4194304
  bf16* Wo_bf  = Wq_bf + 4194304;            // 4194304
  bf16* lat_bf = Wo_bf + 4194304;            // 4096*512
  bf16* q_bf   = lat_bf + 2097152;           // [b,h,s,d]
  bf16* k_bf   = q_bf  + 8388608;            // [b,h,s,d]
  bf16* vT_bf  = k_bf  + 8388608;            // [b,h,d,s]
  bf16* ctx_bf = vT_bf + 8388608;            // [4096, 2048]

  const int TB = 256;
  cvt_f32_to_bf16<<<(8388608 + TB - 1) / TB, TB, 0, stream>>>(x,  x_bf,  8388608);
  cvt_f32_to_bf16<<<(1048576 + TB - 1) / TB, TB, 0, stream>>>(Wd, Wd_bf, 1048576);
  cvt_f32_to_bf16<<<(1048576 + TB - 1) / TB, TB, 0, stream>>>(Wk, Wk_bf, 1048576);
  cvt_f32_to_bf16<<<(1048576 + TB - 1) / TB, TB, 0, stream>>>(Wv, Wv_bf, 1048576);
  cvt_f32_to_bf16<<<(4194304 + TB - 1) / TB, TB, 0, stream>>>(Wq, Wq_bf, 4194304);
  cvt_f32_to_bf16<<<(4194304 + TB - 1) / TB, TB, 0, stream>>>(Wo, Wo_bf, 4194304);

  // latents = x @ Wd^T : [4096, 512]
  gemm_bf16_wmma<0, false><<<dim3(D_LATENT / 64, ROWS / 256), 256, 0, stream>>>(
      x_bf, Wd_bf, lat_bf, ROWS, D_LATENT, D_MODEL);
  // q = x @ Wq^T -> [b,h,s,d]
  gemm_bf16_wmma<1, false><<<dim3(D_MODEL / 64, ROWS / 256), 256, 0, stream>>>(
      x_bf, Wq_bf, q_bf, ROWS, D_MODEL, D_MODEL);
  // k = latents @ Wk^T -> [b,h,s,d]
  gemm_bf16_wmma<1, false><<<dim3(D_MODEL / 64, ROWS / 256), 256, 0, stream>>>(
      lat_bf, Wk_bf, k_bf, ROWS, D_MODEL, D_LATENT);
  // v = latents @ Wv^T -> [b,h,d,s]  (transposed for PV fragment loads)
  gemm_bf16_wmma<2, false><<<dim3(D_MODEL / 64, ROWS / 256), 256, 0, stream>>>(
      lat_bf, Wv_bf, vT_bf, ROWS, D_MODEL, D_LATENT);

  // attention: 4096 waves, 8 per block
  mla_flash_attn<<<512, 256, 0, stream>>>(q_bf, k_bf, vT_bf, ctx_bf);

  // out = ctx @ Wo^T : f32 [4096, 2048]
  gemm_bf16_wmma<0, true><<<dim3(D_MODEL / 64, ROWS / 256), 256, 0, stream>>>(
      ctx_bf, Wo_bf, out, ROWS, D_MODEL, D_MODEL);
}
